// DCN_off_50070728737064
// MI455X (gfx1250) — compile-verified, matrix operand order checked by reading, not run on hardware
//
#include <hip/hip_runtime.h>
#include <hip/hip_bf16.h>
#include <cstddef>

typedef __attribute__((ext_vector_type(16))) _Float16 v16h;
typedef __attribute__((ext_vector_type(8)))  _Float16 v8h;
typedef __attribute__((ext_vector_type(8)))  float    v8f;

union V16H { v16h v; v8h h[2]; };

#define HH 256
#define WW 256
#define NB 2
#define HWSZ (HH*WW)
#define PIX (NB*HWSZ)

// K-index held by (lane, half) in the documented 16-bit 16x32 A/B fragment:
// lanes 0-15: V0-V3 -> K 0..7, V4-V7 -> K 16..23 ; lanes 16-31: +8.
__device__ __forceinline__ int frag_k(int lane, int half) {
  int p = half >> 1, o = half & 1;
  int base = (lane & 16) ? 8 : 0;
  return (p < 4) ? (base + 2 * p + o) : (16 + base + 2 * (p - 4) + o);
}

__device__ __forceinline__ float leaky(float v) { return v >= 0.f ? v : 0.1f * v; }

// ---------------------------------------------------------------------------
// Permute f32 conv weights into WMMA A-fragment order:
// dst layout [COUT_PAD/16][KTOT/32][lane 32][half 16] f16, rows >= COUT are 0.
// korder==0: Kg = tap*CIN + ci (tap-major, for 3x3 convs)
// korder==1: Kg = ci*9 + tap  (natural flatten, for the deform GEMM)
// ---------------------------------------------------------------------------
__global__ void k_prep_weights(const float* __restrict__ w, _Float16* __restrict__ dst,
                               int COUT, int COUT_PAD, int CIN, int KTOT, int korder) {
  int kch = KTOT / 32;
  int total = (COUT_PAD / 16) * kch * 512;
  for (int idx = blockIdx.x * blockDim.x + threadIdx.x; idx < total;
       idx += gridDim.x * blockDim.x) {
    int half    = idx & 15;
    int lane    = (idx >> 4) & 31;
    int kchunk  = (idx >> 9) % kch;
    int co_tile = idx / (512 * kch);
    int Kg = kchunk * 32 + frag_k(lane, half);
    int co = co_tile * 16 + (lane & 15);
    float v = 0.f;
    if (co < COUT) {
      int src;
      if (korder == 0) { int tap = Kg / CIN, ci = Kg % CIN; src = (co * CIN + ci) * 9 + tap; }
      else             { src = co * KTOT + Kg; }
      v = w[src];
    }
    dst[idx] = (_Float16)v;
  }
}

// concat(ali, ref) -> f16 [B][128][H][W]
__global__ void k_prep_in1(const float* __restrict__ ali, const float* __restrict__ ref,
                           _Float16* __restrict__ dst) {
  const int N = NB * 128 * HWSZ;
  for (int i = blockIdx.x * blockDim.x + threadIdx.x; i < N; i += gridDim.x * blockDim.x) {
    int pix = i % HWSZ;
    int c   = (i / HWSZ) % 128;
    int b   = i / (128 * HWSZ);
    float v = (c < 64) ? ali[((size_t)b * 64 + c) * HWSZ + pix]
                       : ref[((size_t)b * 64 + (c - 64)) * HWSZ + pix];
    dst[i] = (_Float16)v;
  }
}

// offset_pre*2 -> f16 channels [0,64) of the 128-channel in4 buffer
__global__ void k_prep_in4a(const float* __restrict__ op, _Float16* __restrict__ in4) {
  const int N = NB * 64 * HWSZ;
  for (int i = blockIdx.x * blockDim.x + threadIdx.x; i < N; i += gridDim.x * blockDim.x) {
    int pix = i % HWSZ;
    int c   = (i / HWSZ) % 64;
    int b   = i / (64 * HWSZ);
    in4[((size_t)b * 128 + c) * HWSZ + pix] = (_Float16)(2.f * op[i]);
  }
}

// ---------------------------------------------------------------------------
// 3x3 SAME conv as implicit GEMM, WMMA f32 <- f16xf16 (16x16x32).
// Block: 256 thr = 8 waves; tile = 64 Cout x 32 pixels (4x2 waves of 16x16).
// grid = (PIX/32, COUT_PAD/64). A frags streamed from prepped global buffer;
// B tile = 32-channel x 3x34 input halo staged in LDS per channel chunk.
// ---------------------------------------------------------------------------
template <int CIN, int COUT, int COUT_PAD, int ACT>
__global__ void k_conv3x3_wmma(const _Float16* __restrict__ in,
                               const _Float16* __restrict__ wfrag,
                               const float* __restrict__ bias,
                               _Float16* __restrict__ out_f16, int oc_total, int oc_off,
                               float* __restrict__ out_f32) {
  constexpr int KCH = CIN * 9 / 32;           // K chunks of 32
  __shared__ __align__(16) _Float16 sB[3 * 34 * 32];

  const int tid = threadIdx.x;
  const int wave = tid >> 5, lane = tid & 31;
  const int mb = wave & 3, nb = wave >> 2;    // 4 x 2 wave grid
  const int tile_pix = blockIdx.x * 32;
  const int b = tile_pix / HWSZ;
  const int rem = tile_pix % HWSZ;
  const int y = rem / WW, x0 = rem % WW;
  const int co_tile = blockIdx.y * 4 + mb;    // 16-row Cout tile index
  const int col = lane & 15;
  const int cibase = (lane & 16) ? 8 : 0;

  v8f acc = {0.f, 0.f, 0.f, 0.f, 0.f, 0.f, 0.f, 0.f};

  for (int cc = 0; cc < CIN / 32; ++cc) {
    __syncthreads();
    // stage in[b][cc*32 .. +31][y-1..y+1][x0-1..x0+32] as LDS [row][x][ci]
    for (int idx = tid; idx < 3 * 34 * 32; idx += 256) {
      int xx = idx % 34;
      int ci = (idx / 34) % 32;
      int r  = idx / (34 * 32);
      int yy = y + r - 1;
      int xg = x0 - 1 + xx;
      _Float16 v = (_Float16)0.f;
      if (yy >= 0 && yy < HH && xg >= 0 && xg < WW)
        v = in[(((size_t)b * CIN + cc * 32 + ci) * HH + yy) * WW + xg];
      sB[(r * 34 + xx) * 32 + ci] = v;
    }
    __syncthreads();
#pragma unroll
    for (int tap = 0; tap < 9; ++tap) {
      const int ky = tap / 3, kx = tap % 3;
      const int kchunk = tap * (CIN / 32) + cc;
      const _Float16* ap = wfrag + ((size_t)(co_tile * KCH + kchunk) * 32 + lane) * 16;
      __builtin_prefetch(ap + 512, 0, 1);     // next chunk's A, global_prefetch_b8
      V16H A, Bf;
      A.h[0] = *(const v8h*)ap;
      A.h[1] = *(const v8h*)(ap + 8);
      const int xxi = nb * 16 + col + kx;     // = (px + kx - 1) - (x0 - 1)
      const _Float16* bp = &sB[(ky * 34 + xxi) * 32 + cibase];
      Bf.h[0] = *(const v8h*)bp;
      Bf.h[1] = *(const v8h*)(bp + 16);
      acc = __builtin_amdgcn_wmma_f32_16x16x32_f16(false, A.v, false, Bf.v,
                                                   (short)0, acc, false, false);
    }
  }

  // C/D layout: VGPR r -> (M = r + (lane>=16 ? 8:0), N = lane&15)
  const int co0 = co_tile * 16 + ((lane & 16) ? 8 : 0);
  const int pxe = x0 + nb * 16 + col;
#pragma unroll
  for (int r = 0; r < 8; ++r) {
    int co = co0 + r;
    if (co < COUT) {
      float v = acc[r] + bias[co];
      if (ACT) v = leaky(v);
      if (out_f16)
        out_f16[(((size_t)b * oc_total + oc_off + co) * HH + y) * WW + pxe] = (_Float16)v;
      if (out_f32)
        out_f32[(((size_t)b * COUT + co) * HH + y) * WW + pxe] = v;
    }
  }
}

// ---------------------------------------------------------------------------
// Deformable conv: bilinear-gather + mask into an LDS value tile
// (32 px x 576 K, stored in fragment order), then 18 WMMAs per wave for the
// 64x32 output tile. om = [dy,dx interleaved (144ch) | mask_raw (72ch)].
// ---------------------------------------------------------------------------
__global__ void k_deform_wmma(const _Float16* __restrict__ xf,   // [B][64][H][W] f16
                              const float* __restrict__ om,      // [B][216][H][W]
                              const _Float16* __restrict__ wfrag,// [4][18][32][16]
                              const float* __restrict__ bias,
                              float* __restrict__ out) {         // [B][64][H][W]
  __shared__ __align__(16) _Float16 sV[2 * 18 * 32 * 16];        // 36 KB

  const int tid = threadIdx.x;
  const int wave = tid >> 5, lane = tid & 31;
  const int mb = wave & 3, nb = wave >> 2;
  const int tile_pix = blockIdx.x * 32;
  const int b = tile_pix / HWSZ;
  const int rem = tile_pix % HWSZ;
  const int y = rem / WW, x0 = rem % WW;

  // Fill value tile: 18432 entries / 256 threads = 72 each (exact).
#pragma unroll 4
  for (int it = 0; it < 72; ++it) {
    const int idx = it * 256 + tid;
    const int half   = idx & 15;
    const int lane_f = (idx >> 4) & 31;
    const int kchunk = (idx >> 9) % 18;
    const int nbf    = idx / (18 * 512);
    const int Kg = kchunk * 32 + frag_k(lane_f, half);
    const int i = Kg / 9, kk = Kg % 9;
    const int g = i >> 3;                     // 8 channels per group
    const int x = x0 + nbf * 16 + (lane_f & 15);
    const int ch = g * 9 + kk;
    const size_t sp = (size_t)y * WW + x;
    const float dy = om[((size_t)b * 216 + 2 * ch) * HWSZ + sp];
    const float dx = om[((size_t)b * 216 + 2 * ch + 1) * HWSZ + sp];
    const float mr = om[((size_t)b * 216 + 144 + ch) * HWSZ + sp];
    const float m  = 1.f / (1.f + __expf(-mr));
    const float py = (float)y + (float)(kk / 3 - 1) + dy;
    const float px = (float)x + (float)(kk % 3 - 1) + dx;
    const float fy = floorf(py), fx = floorf(px);
    const int y0 = (int)fy, x0i = (int)fx;
    const float ay = py - fy, ax = px - fx;
    const _Float16* xc = xf + ((size_t)b * 64 + i) * HWSZ;
    auto g4 = [&](int yy, int xx) -> float {
      if (yy < 0 || yy >= HH || xx < 0 || xx >= WW) return 0.f;
      return (float)xc[(size_t)yy * WW + xx];
    };
    float v = g4(y0, x0i)     * (1.f - ay) * (1.f - ax)
            + g4(y0, x0i + 1) * (1.f - ay) * ax
            + g4(y0 + 1, x0i) * ay * (1.f - ax)
            + g4(y0 + 1, x0i + 1) * ay * ax;
    sV[idx] = (_Float16)(v * m);
  }
  __syncthreads();

  v8f acc = {0.f, 0.f, 0.f, 0.f, 0.f, 0.f, 0.f, 0.f};
#pragma unroll
  for (int kchunk = 0; kchunk < 18; ++kchunk) {
    const _Float16* ap = wfrag + ((size_t)(mb * 18 + kchunk) * 32 + lane) * 16;
    const _Float16* bp = &sV[((size_t)(nb * 18 + kchunk) * 32 + lane) * 16];
    V16H A, Bf;
    A.h[0]  = *(const v8h*)ap;  A.h[1]  = *(const v8h*)(ap + 8);
    Bf.h[0] = *(const v8h*)bp;  Bf.h[1] = *(const v8h*)(bp + 8);
    acc = __builtin_amdgcn_wmma_f32_16x16x32_f16(false, A.v, false, Bf.v,
                                                 (short)0, acc, false, false);
  }

  const int co0 = mb * 16 + ((lane & 16) ? 8 : 0);
  const int pxe = x0 + nb * 16 + (lane & 15);
#pragma unroll
  for (int r = 0; r < 8; ++r) {
    const int co = co0 + r;
    out[(((size_t)b * 64 + co) * HH + y) * WW + pxe] = leaky(acc[r] + bias[co]);
  }
}

// ---------------------------------------------------------------------------
extern "C" void kernel_launch(void* const* d_in, const int* in_sizes, int n_in,
                              void* d_out, int out_size, void* d_ws, size_t ws_size,
                              hipStream_t stream) {
  (void)in_sizes; (void)n_in; (void)out_size; (void)ws_size;
  const float* ali       = (const float*)d_in[0];
  const float* ref       = (const float*)d_in[1];
  const float* off_pre   = (const float*)d_in[2];
  const float* conv_w    = (const float*)d_in[3];
  const float* conv_b    = (const float*)d_in[4];
  const float* off1_w    = (const float*)d_in[5];
  const float* off1_b    = (const float*)d_in[6];
  const float* x_w       = (const float*)d_in[7];
  const float* x_b       = (const float*)d_in[8];
  const float* off2_w    = (const float*)d_in[9];
  const float* off2_b    = (const float*)d_in[10];
  const float* dcn_om_w  = (const float*)d_in[11];
  const float* dcn_om_b  = (const float*)d_in[12];
  const float* dcn_w     = (const float*)d_in[13];
  const float* dcn_b     = (const float*)d_in[14];

  float* out_main = (float*)d_out;                       // output  [2][64][256][256]
  float* out_off  = (float*)d_out + (size_t)NB * 64 * HWSZ; // offset

  char* ws = (char*)d_ws;
  size_t o = 0;
  auto alloc = [&](size_t bytes) { void* p = ws + o; o += bytes; return p; };
  _Float16* wA1   = (_Float16*)alloc(294912);   // 128x(128*9) padded 128
  _Float16* wA2   = (_Float16*)alloc(147456);   // 64x(128*9)
  _Float16* wA3   = (_Float16*)alloc(147456);
  _Float16* wA4   = (_Float16*)alloc(147456);
  _Float16* wA5   = (_Float16*)alloc(294912);   // 216->256 x (64*9)
  _Float16* wA6   = (_Float16*)alloc(73728);    // 64 x 576 (deform)
  _Float16* in1   = (_Float16*)alloc((size_t)NB * 128 * HWSZ * 2); // reused as in4
  _Float16* tbuf  = (_Float16*)alloc((size_t)NB * 128 * HWSZ * 2);
  _Float16* xbuf  = (_Float16*)alloc((size_t)NB * 64 * HWSZ * 2);
  _Float16* offb  = (_Float16*)alloc((size_t)NB * 64 * HWSZ * 2);
  float*    omb   = (float*)   alloc((size_t)NB * 216 * HWSZ * 4);
  _Float16* in4   = in1;                        // in1 dead after conv1

  // weight permutation (independent, tiny)
  k_prep_weights<<<288, 256, 0, stream>>>(conv_w,   wA1, 128, 128, 128, 1152, 0);
  k_prep_weights<<<288, 256, 0, stream>>>(off1_w,   wA2,  64,  64, 128, 1152, 0);
  k_prep_weights<<<288, 256, 0, stream>>>(x_w,      wA3,  64,  64, 128, 1152, 0);
  k_prep_weights<<<288, 256, 0, stream>>>(off2_w,   wA4,  64,  64, 128, 1152, 0);
  k_prep_weights<<<288, 256, 0, stream>>>(dcn_om_w, wA5, 216, 256,  64,  576, 0);
  k_prep_weights<<<144, 256, 0, stream>>>(dcn_w,    wA6,  64,  64,  64,  576, 1);

  k_prep_in1<<<16384, 256, 0, stream>>>(ali, ref, in1);

  const dim3 blk(256);
  // t = leaky(conv(concat(ali,ref)))          128 -> 128
  k_conv3x3_wmma<128, 128, 128, 1><<<dim3(PIX / 32, 2), blk, 0, stream>>>(
      in1, wA1, conv_b, tbuf, 128, 0, nullptr);
  // in4[0:64] = 2*offset_pre (in1 buffer now free)
  k_prep_in4a<<<8192, 256, 0, stream>>>(off_pre, in4);
  // off_feat -> in4[64:128]                   128 -> 64
  k_conv3x3_wmma<128, 64, 64, 1><<<dim3(PIX / 32, 1), blk, 0, stream>>>(
      tbuf, wA2, off1_b, in4, 128, 64, nullptr);
  // x                                         128 -> 64
  k_conv3x3_wmma<128, 64, 64, 1><<<dim3(PIX / 32, 1), blk, 0, stream>>>(
      tbuf, wA3, x_b, xbuf, 64, 0, nullptr);
  // offset (f16 for next conv, f32 to d_out)  128 -> 64
  k_conv3x3_wmma<128, 64, 64, 1><<<dim3(PIX / 32, 1), blk, 0, stream>>>(
      in4, wA4, off2_b, offb, 64, 0, out_off);
  // om (no activation)                        64 -> 216 (padded 256)
  k_conv3x3_wmma<64, 216, 256, 0><<<dim3(PIX / 32, 4), blk, 0, stream>>>(
      offb, wA5, dcn_om_b, nullptr, 0, 0, omb);
  // deformable conv -> output
  k_deform_wmma<<<PIX / 32, blk, 0, stream>>>(xbuf, omb, wA6, dcn_b, out_main);
}
